// SlicedReLUBumpSelfAttention_32023276159753
// MI455X (gfx1250) — matrix operand order repaired and checked by reading, not compiled, not run on hardware
//
#include <hip/hip_runtime.h>
#include <hip/hip_bf16.h>
#include <math.h>
#include <stdint.h>

typedef __attribute__((ext_vector_type(16))) __bf16 bf16x16;
typedef __attribute__((ext_vector_type(8)))  __bf16 bf16x8;
typedef __attribute__((ext_vector_type(4)))  __bf16 bf16x4;
typedef __attribute__((ext_vector_type(8)))  float  f32x8;

#define B_  4
#define T_  2048
#define HS_ 1024
#define H_  16
#define D_  64
#define L2_ 4096   /* 2T */
#define BH_ 64     /* B*H */

// ---------------------------------------------------------------------------
// Async global->LDS helpers (gfx1250): tracked by ASYNCcnt.
// LDS destination address = low 32 bits of the flat shared-memory pointer
// (flat LDS addressing truncates to addr[31:0] per CDNA5 ISA 10.2).
// ---------------------------------------------------------------------------
__device__ __forceinline__ void async_ld16(uint32_t lds_addr, const void* gaddr) {
  asm volatile("global_load_async_to_lds_b128 %0, %1, off"
               :: "v"(lds_addr), "v"(gaddr) : "memory");
}
__device__ __forceinline__ uint32_t lds_lo32(const void* p) {
  return (uint32_t)(uintptr_t)p;
}

// ---------------------------------------------------------------------------
// Pre-pass: split fp32 array into hi/lo bf16 arrays (x == hi + lo to ~2^-16).
// ---------------------------------------------------------------------------
__global__ __launch_bounds__(256)
void split_bf16_kernel(const float* __restrict__ src, __bf16* __restrict__ hi,
                       __bf16* __restrict__ lo, int n4) {
  int i = blockIdx.x * 256 + threadIdx.x;
  if (i < n4) {
    float4 x = ((const float4*)src)[i];
    bf16x4 h, l;
    h[0] = (__bf16)x.x; l[0] = (__bf16)(x.x - (float)h[0]);
    h[1] = (__bf16)x.y; l[1] = (__bf16)(x.y - (float)h[1]);
    h[2] = (__bf16)x.z; l[2] = (__bf16)(x.z - (float)h[2]);
    h[3] = (__bf16)x.w; l[3] = (__bf16)(x.w - (float)h[3]);
    ((bf16x4*)hi)[i] = h;
    ((bf16x4*)lo)[i] = l;
  }
}

// ---------------------------------------------------------------------------
// QKV GEMM: C[r][c] = sum_k X[r][k] * W[c][k] + bias[c]
// Split-bf16 (hi+lo): 3x v_wmma_f32_16x16x32_bf16 per 16x16 tile per k-step.
// Double-buffered LDS staged with global_load_async_to_lds_b128.
// Output scattered into (B,H,T,D) layout: r=b*T+t, c=h*D+d.
// ---------------------------------------------------------------------------
#define BM 128
#define BN 64
#define BK 32
#define LDA 40   // padded LDS row stride (halves); 80B = multiple of 16B
#define NKB (HS_ / BK)

__device__ __forceinline__ void stage_tile(
    const __bf16* __restrict__ Xhi, const __bf16* __restrict__ Xlo,
    const __bf16* __restrict__ Whi, const __bf16* __restrict__ Wlo,
    __bf16* sAhi, __bf16* sAlo, __bf16* sBhi, __bf16* sBlo,
    size_t rowA0, size_t rowB0, int k0, int tid) {
  // X tile: 128 rows x 32 halves = 512 vectors of 8 halves (2 per thread).
#pragma unroll
  for (int v = tid; v < 512; v += 256) {
    int r = v >> 2, kk = (v & 3) * 8;
    size_t go = (rowA0 + r) * HS_ + (size_t)(k0 + kk);
    uint32_t lo_ = (uint32_t)(r * LDA + kk);
    async_ld16(lds_lo32(sAhi + lo_), Xhi + go);
    async_ld16(lds_lo32(sAlo + lo_), Xlo + go);
  }
  // W tile: 64 rows x 32 halves = 256 vectors (1 per thread).
#pragma unroll
  for (int v = tid; v < 256; v += 256) {
    int n = v >> 2, kk = (v & 3) * 8;
    size_t go = (rowB0 + n) * HS_ + (size_t)(k0 + kk);
    uint32_t lo_ = (uint32_t)(n * LDA + kk);
    async_ld16(lds_lo32(sBhi + lo_), Whi + go);
    async_ld16(lds_lo32(sBlo + lo_), Wlo + go);
  }
}

__global__ __launch_bounds__(256)
void qkv_gemm_kernel(const __bf16* __restrict__ Xhi, const __bf16* __restrict__ Xlo,
                     const __bf16* __restrict__ Whi, const __bf16* __restrict__ Wlo,
                     const float* __restrict__ bias, float* __restrict__ out) {
  __shared__ __attribute__((aligned(16))) __bf16 sAhi[2][BM * LDA];
  __shared__ __attribute__((aligned(16))) __bf16 sAlo[2][BM * LDA];
  __shared__ __attribute__((aligned(16))) __bf16 sBhi[2][BN * LDA];
  __shared__ __attribute__((aligned(16))) __bf16 sBlo[2][BN * LDA];

  const int tid   = threadIdx.x;
  const int lane  = tid & 31;
  const int wave  = tid >> 5;        // 8 waves
  const int waveM = wave >> 1;       // 0..3  -> 32-row strip
  const int waveN = wave & 1;        // 0..1  -> 32-col strip
  const int half  = lane >> 4;       // lane group
  const int l16   = lane & 15;
  const int bm = blockIdx.y, bn = blockIdx.x;
  const size_t rowA0 = (size_t)bm * BM;
  const size_t rowB0 = (size_t)bn * BN;

  f32x8 acc[2][2];
#pragma unroll
  for (int mt = 0; mt < 2; ++mt)
#pragma unroll
    for (int nt = 0; nt < 2; ++nt) acc[mt][nt] = (f32x8){0,0,0,0,0,0,0,0};

  // Preload first tile into buffer 0.
  stage_tile(Xhi, Xlo, Whi, Wlo, sAhi[0], sAlo[0], sBhi[0], sBlo[0],
             rowA0, rowB0, 0, tid);

  for (int kb = 0; kb < NKB; ++kb) {
    const int cur = kb & 1;
    const bool hasNext = (kb + 1) < NKB;
    if (hasNext) {
      stage_tile(Xhi, Xlo, Whi, Wlo, sAhi[cur ^ 1], sAlo[cur ^ 1],
                 sBhi[cur ^ 1], sBlo[cur ^ 1], rowA0, rowB0, (kb + 1) * BK, tid);
      // 6 new async loads outstanding -> waiting to <=6 drains the previous tile.
      asm volatile("s_wait_asynccnt 0x6" ::: "memory");
    } else {
      asm volatile("s_wait_asynccnt 0x0" ::: "memory");
    }
    __syncthreads();

    bf16x16 ahi[2], alo[2], bhi[2], blo[2];
#pragma unroll
    for (int mt = 0; mt < 2; ++mt) {
      int row = waveM * 32 + mt * 16 + l16;
      const bf16x8* p0 = (const bf16x8*)&sAhi[cur][row * LDA + half * 8];
      const bf16x8* p1 = (const bf16x8*)&sAhi[cur][row * LDA + 16 + half * 8];
      ahi[mt] = __builtin_shufflevector(*p0, *p1, 0,1,2,3,4,5,6,7,8,9,10,11,12,13,14,15);
      const bf16x8* q0 = (const bf16x8*)&sAlo[cur][row * LDA + half * 8];
      const bf16x8* q1 = (const bf16x8*)&sAlo[cur][row * LDA + 16 + half * 8];
      alo[mt] = __builtin_shufflevector(*q0, *q1, 0,1,2,3,4,5,6,7,8,9,10,11,12,13,14,15);
    }
#pragma unroll
    for (int nt = 0; nt < 2; ++nt) {
      int n = waveN * 32 + nt * 16 + l16;
      const bf16x8* p0 = (const bf16x8*)&sBhi[cur][n * LDA + half * 8];
      const bf16x8* p1 = (const bf16x8*)&sBhi[cur][n * LDA + 16 + half * 8];
      bhi[nt] = __builtin_shufflevector(*p0, *p1, 0,1,2,3,4,5,6,7,8,9,10,11,12,13,14,15);
      const bf16x8* q0 = (const bf16x8*)&sBlo[cur][n * LDA + half * 8];
      const bf16x8* q1 = (const bf16x8*)&sBlo[cur][n * LDA + 16 + half * 8];
      blo[nt] = __builtin_shufflevector(*q0, *q1, 0,1,2,3,4,5,6,7,8,9,10,11,12,13,14,15);
    }

#pragma unroll
    for (int mt = 0; mt < 2; ++mt)
#pragma unroll
      for (int nt = 0; nt < 2; ++nt) {
        acc[mt][nt] = __builtin_amdgcn_wmma_f32_16x16x32_bf16(
            false, ahi[mt], false, bhi[nt], (short)0, acc[mt][nt], false, false);
        acc[mt][nt] = __builtin_amdgcn_wmma_f32_16x16x32_bf16(
            false, ahi[mt], false, blo[nt], (short)0, acc[mt][nt], false, false);
        acc[mt][nt] = __builtin_amdgcn_wmma_f32_16x16x32_bf16(
            false, alo[mt], false, bhi[nt], (short)0, acc[mt][nt], false, false);
      }
    __syncthreads();  // all reads of buf[cur] done before it is restaged
  }

  // Epilogue: add bias, scatter to (B,H,T,D).
#pragma unroll
  for (int mt = 0; mt < 2; ++mt)
#pragma unroll
    for (int nt = 0; nt < 2; ++nt) {
      int n = bn * BN + waveN * 32 + nt * 16 + l16;
      int h = n >> 6, d = n & 63;
      float bv = bias[n];
#pragma unroll
      for (int e = 0; e < 8; ++e) {
        int m = waveM * 32 + mt * 16 + half * 8 + e;
        size_t R = rowA0 + m;
        int bb = (int)(R >> 11);
        int t  = (int)(R & 2047);
        out[(((size_t)bb * H_ + h) * T_ + t) * D_ + d] = acc[mt][nt][e] + bv;
      }
    }
}

// ---------------------------------------------------------------------------
// Projection: Z row = (q|k).reshape(B,T,1024) @ Wproj.T  (N=16, fp32 exact)
// Writes Z[(b*H+h)*4096 + zoff + t].
// ---------------------------------------------------------------------------
__global__ __launch_bounds__(256)
void proj_kernel(const float* __restrict__ Q, const float* __restrict__ Wp,
                 float* __restrict__ Z, int zoff) {
  int tid = threadIdx.x;
  int r = blockIdx.x * 16 + (tid >> 4);
  int h = tid & 15;
  const float4* row = (const float4*)(Q + (size_t)r * HS_);
  const float4* w   = (const float4*)(Wp + (size_t)h * HS_);
  float acc = 0.f;
#pragma unroll 4
  for (int k = 0; k < HS_ / 4; ++k) {
    float4 a = row[k], b = w[k];
    acc += a.x * b.x + a.y * b.y + a.z * b.z + a.w * b.w;
  }
  int b = r >> 11, t = r & 2047;
  Z[((size_t)(b * H_ + h)) * L2_ + zoff + t] = acc;
}

// ---------------------------------------------------------------------------
// Bitonic sort of 4096 (z, idx) pairs per (b,h) in LDS.
// ---------------------------------------------------------------------------
__global__ __launch_bounds__(256)
void sort_kernel(const float* __restrict__ Z, float* __restrict__ zs,
                 int* __restrict__ idx) {
  __shared__ float key[L2_];
  __shared__ int   val[L2_];
  int bh = blockIdx.x, tid = threadIdx.x;
  for (int i = tid; i < L2_; i += 256) { key[i] = Z[(size_t)bh * L2_ + i]; val[i] = i; }
  __syncthreads();
  for (int k = 2; k <= L2_; k <<= 1) {
    for (int j = k >> 1; j > 0; j >>= 1) {
      for (int t = tid; t < L2_; t += 256) {
        int p = t ^ j;
        if (p > t) {
          float a = key[t], b = key[p];
          bool up = ((t & k) == 0);
          if ((a > b) == up) {
            key[t] = b; key[p] = a;
            int va = val[t]; val[t] = val[p]; val[p] = va;
          }
        }
      }
      __syncthreads();
    }
  }
  for (int i = tid; i < L2_; i += 256) {
    zs[(size_t)bh * L2_ + i] = key[i];
    idx[(size_t)bh * L2_ + i] = val[i];
  }
}

// ---------------------------------------------------------------------------
// Chunked prefix sums of v_sorted and z*v_sorted: 64 chunks of 64 along L.
// ---------------------------------------------------------------------------
__global__ __launch_bounds__(256)
void scan_part_kernel(const float* __restrict__ zs, const int* __restrict__ idx,
                      const float* __restrict__ v, float* __restrict__ pvp,
                      float* __restrict__ pzvp) {
  int tid = threadIdx.x;
  int d = tid & 63, cl = tid >> 6;
  int bh = blockIdx.x;
  int chunk = blockIdx.y * 4 + cl;
  const float* vb = v + (size_t)bh * T_ * D_ + d;
  float sv = 0.f, szv = 0.f;
  int base = chunk * 64;
  for (int ii = 0; ii < 64; ++ii) {
    int i = base + ii;
    int j = idx[(size_t)bh * L2_ + i];
    float z = zs[(size_t)bh * L2_ + i];
    float vj = (j < T_) ? vb[(size_t)j * D_] : 0.f;
    sv += vj; szv += z * vj;
  }
  pvp [((size_t)bh * 64 + chunk) * 64 + d] = sv;
  pzvp[((size_t)bh * 64 + chunk) * 64 + d] = szv;
}

__global__ __launch_bounds__(64)
void scan_offsets_kernel(float* __restrict__ pvp, float* __restrict__ pzvp) {
  int bh = blockIdx.x, d = threadIdx.x;
  float rv = 0.f, rz = 0.f;
  for (int c = 0; c < 64; ++c) {
    size_t o = ((size_t)bh * 64 + c) * 64 + d;
    float a = pvp[o];  pvp[o]  = rv; rv += a;
    float b = pzvp[o]; pzvp[o] = rz; rz += b;
  }
}

__global__ __launch_bounds__(256)
void scan_write_kernel(const float* __restrict__ zs, const int* __restrict__ idx,
                       const float* __restrict__ v, const float* __restrict__ pvp,
                       const float* __restrict__ pzvp, float* __restrict__ Pv,
                       float* __restrict__ Pzv) {
  int tid = threadIdx.x;
  int d = tid & 63, cl = tid >> 6;
  int bh = blockIdx.x;
  int chunk = blockIdx.y * 4 + cl;
  const float* vb = v + (size_t)bh * T_ * D_ + d;
  float sv  = pvp [((size_t)bh * 64 + chunk) * 64 + d];
  float szv = pzvp[((size_t)bh * 64 + chunk) * 64 + d];
  if (chunk == 0) {
    Pv [(size_t)bh * (L2_ + 1) * D_ + d] = 0.f;
    Pzv[(size_t)bh * (L2_ + 1) * D_ + d] = 0.f;
  }
  int base = chunk * 64;
  for (int ii = 0; ii < 64; ++ii) {
    int i = base + ii;
    int j = idx[(size_t)bh * L2_ + i];
    float z = zs[(size_t)bh * L2_ + i];
    float vj = (j < T_) ? vb[(size_t)j * D_] : 0.f;
    sv += vj; szv += z * vj;
    Pv [((size_t)bh * (L2_ + 1) + i + 1) * D_ + d] = sv;
    Pzv[((size_t)bh * (L2_ + 1) + i + 1) * D_ + d] = szv;
  }
}

// ---------------------------------------------------------------------------
// Combine: searchsorted in LDS, bump formula, scatter q-slots to output.
// ---------------------------------------------------------------------------
__global__ __launch_bounds__(256)
void combine_kernel(const float* __restrict__ zs, const int* __restrict__ idx,
                    const float* __restrict__ Pv, const float* __restrict__ Pzv,
                    const float* __restrict__ log_bw, float* __restrict__ out) {
  __shared__ float zl[L2_];
  __shared__ int sL[256], sR[256], sG[256];
  int bh = blockIdx.x, tile = blockIdx.y, tid = threadIdx.x;
  int b = bh >> 4, h = bh & 15;
  for (int i = tid; i < L2_; i += 256) zl[i] = zs[(size_t)bh * L2_ + i];
  __syncthreads();
  float x = log_bw[h];
  float bw = ((x > 20.f) ? x : log1pf(expf(x))) + 1e-4f;
  {
    int i0 = tile * 256 + tid;
    float q = zl[i0] - bw;
    int lo = 0, hi = L2_;
    while (lo < hi) { int mid = (lo + hi) >> 1; if (zl[mid] < q) lo = mid + 1; else hi = mid; }
    sL[tid] = lo;
    float q2 = zl[i0] + bw;
    lo = 0; hi = L2_;
    while (lo < hi) { int mid = (lo + hi) >> 1; if (zl[mid] <= q2) lo = mid + 1; else hi = mid; }
    sR[tid] = lo;
    sG[tid] = idx[(size_t)bh * L2_ + i0];
  }
  __syncthreads();
  int d = tid & 63, sub = tid >> 6;
  float invbw = 1.f / bw;
  const float* pv = Pv  + (size_t)bh * (L2_ + 1) * D_ + d;
  const float* pz = Pzv + (size_t)bh * (L2_ + 1) * D_ + d;
  for (int pass = 0; pass < 64; ++pass) {
    int il = pass * 4 + sub;
    int i = tile * 256 + il;
    int Li = sL[il], Ri = sR[il], g = sG[il];
    float zi = zl[i];
    float pv1 = pv[(size_t)(i + 1) * D_], pvL = pv[(size_t)Li * D_], pvR = pv[(size_t)Ri * D_];
    float pz1 = pz[(size_t)(i + 1) * D_], pzL = pz[(size_t)Li * D_], pzR = pz[(size_t)Ri * D_];
    float svl = pv1 - pvL, szl = pz1 - pzL;
    float svr = pvR - pv1, szr = pzR - pz1;
    float zob = zi * invbw;
    float o = svl * (1.f - zob) + szl * invbw + svr * (1.f + zob) - szr * invbw;
    if (g >= T_) {
      int t = g - T_;
      out[((size_t)b * T_ + t) * (H_ * D_) + h * D_ + d] = o * (1.f / (float)T_);
    }
  }
}

// ---------------------------------------------------------------------------
extern "C" void kernel_launch(void* const* d_in, const int* in_sizes, int n_in,
                              void* d_out, int out_size, void* d_ws, size_t ws_size,
                              hipStream_t stream) {
  (void)in_sizes; (void)n_in; (void)out_size; (void)ws_size;
  const float* X   = (const float*)d_in[0];
  const float* Wq  = (const float*)d_in[1];
  const float* bq  = (const float*)d_in[2];
  const float* Wk  = (const float*)d_in[3];
  const float* bk  = (const float*)d_in[4];
  const float* Wv  = (const float*)d_in[5];
  const float* bv  = (const float*)d_in[6];
  const float* Wp  = (const float*)d_in[7];
  const float* lbw = (const float*)d_in[8];
  float* out = (float*)d_out;

  char* ws = (char*)d_ws;
  float*  q_buf = (float*)(ws + 0ull);          // (B,H,T,D)   33,554,432 B
  float*  k_buf = (float*)(ws + 33554432ull);   //             33,554,432 B
  float*  v_buf = (float*)(ws + 67108864ull);   //             33,554,432 B
  float*  Z     = (float*)(ws + 100663296ull);  // (BH,4096)    1,048,576 B
  float*  zsrt  = (float*)(ws + 101711872ull);  //              1,048,576 B
  int*    idx   = (int*)  (ws + 102760448ull);  //              1,048,576 B
  float*  Pv    = (float*)(ws + 103809024ull);  // (BH,4097,64) 67,125,248 B
  float*  Pzv   = (float*)(ws + 170934272ull);  //             67,125,248 B
  float*  pvp   = (float*)(ws + 238059520ull);  //              1,048,576 B
  float*  pzvp  = (float*)(ws + 239108096ull);  //              1,048,576 B
  __bf16* Xhi   = (__bf16*)(ws + 240156672ull); //             16,777,216 B
  __bf16* Xlo   = (__bf16*)(ws + 256933888ull); //             16,777,216 B
  __bf16* Wsp   = (__bf16*)(ws + 273711104ull); // 6 x 2,097,152 B (hi/lo per W)
  __bf16* Whi_q = Wsp + 0 * 1048576;
  __bf16* Wlo_q = Wsp + 1 * 1048576;
  __bf16* Whi_k = Wsp + 2 * 1048576;
  __bf16* Wlo_k = Wsp + 3 * 1048576;
  __bf16* Whi_v = Wsp + 4 * 1048576;
  __bf16* Wlo_v = Wsp + 5 * 1048576;

  // ---- precision-split pre-pass ----
  split_bf16_kernel<<<(B_*T_*HS_/4 + 255) / 256, 256, 0, stream>>>(X,  Xhi,   Xlo,   B_*T_*HS_/4);
  split_bf16_kernel<<<(HS_*HS_/4   + 255) / 256, 256, 0, stream>>>(Wq, Whi_q, Wlo_q, HS_*HS_/4);
  split_bf16_kernel<<<(HS_*HS_/4   + 255) / 256, 256, 0, stream>>>(Wk, Whi_k, Wlo_k, HS_*HS_/4);
  split_bf16_kernel<<<(HS_*HS_/4   + 255) / 256, 256, 0, stream>>>(Wv, Whi_v, Wlo_v, HS_*HS_/4);

  // ---- QKV GEMMs (WMMA, async-staged, double-buffered) ----
  dim3 gGemm(HS_ / BN, (B_ * T_) / BM);
  qkv_gemm_kernel<<<gGemm, 256, 0, stream>>>(Xhi, Xlo, Whi_q, Wlo_q, bq, q_buf);
  qkv_gemm_kernel<<<gGemm, 256, 0, stream>>>(Xhi, Xlo, Whi_k, Wlo_k, bk, k_buf);
  qkv_gemm_kernel<<<gGemm, 256, 0, stream>>>(Xhi, Xlo, Whi_v, Wlo_v, bv, v_buf);

  // ---- scalar projections into Z ----
  proj_kernel<<<(B_ * T_) / 16, 256, 0, stream>>>(k_buf, Wp, Z, 0);
  proj_kernel<<<(B_ * T_) / 16, 256, 0, stream>>>(q_buf, Wp, Z, T_);

  // ---- per-(b,h) sort ----
  sort_kernel<<<BH_, 256, 0, stream>>>(Z, zsrt, idx);

  // ---- prefix sums ----
  scan_part_kernel<<<dim3(BH_, 16), 256, 0, stream>>>(zsrt, idx, v_buf, pvp, pzvp);
  scan_offsets_kernel<<<BH_, 64, 0, stream>>>(pvp, pzvp);
  scan_write_kernel<<<dim3(BH_, 16), 256, 0, stream>>>(zsrt, idx, v_buf, pvp, pzvp, Pv, Pzv);

  // ---- bump combine + scatter to output ----
  combine_kernel<<<dim3(BH_, 16), 256, 0, stream>>>(zsrt, idx, Pv, Pzv, lbw, out);
}
